// GCN_15023795602156
// MI455X (gfx1250) — matrix operand order repaired and checked by reading, not compiled, not run on hardware
//
#include <hip/hip_runtime.h>
#include <hip/hip_bf16.h>

typedef __attribute__((ext_vector_type(16))) __bf16 v16bf;
typedef __attribute__((ext_vector_type(8)))  float  v8f;

#define DIMF 512
#define HIDF 16
#define CF   64

__device__ __forceinline__ unsigned mix32(unsigned x) {
  x ^= x >> 16; x *= 0x7feb352dU; x ^= x >> 15; x *= 0x846ca68bU; x ^= x >> 16;
  return x;
}

// ---------------- degree / normalization ----------------
__global__ void k_deg_init(float* dinv, int n) {
  int i = blockIdx.x * blockDim.x + threadIdx.x;
  if (i < n) dinv[i] = 1.0f;                 // self-loop contributes 1 to degree
}

__global__ void k_deg_accum(const int* __restrict__ col, float* dinv, int E) {
  int e = blockIdx.x * blockDim.x + threadIdx.x;
  if (e < E) atomicAdd(&dinv[col[e]], 1.0f);
}

__global__ void k_rsqrt(float* dinv, int n) {
  int i = blockIdx.x * blockDim.x + threadIdx.x;
  if (i < n) dinv[i] = rsqrtf(dinv[i]);      // deg >= 1 always (self-loops)
}

// ---------------- GEMM1: h1[N,16] = x[N,512] @ W1[512,16]  (bf16 WMMA) ----------------
// One wave handles one 16-row tile; 16 K-steps of v_wmma_f32_16x16x32_bf16.
__global__ void k_gemm1_wmma(const float* __restrict__ x, const float* __restrict__ W1,
                             float* __restrict__ h1, int n) {
  const int gwave = (int)((blockIdx.x * blockDim.x + threadIdx.x) >> 5);
  const int lane  = (int)(threadIdx.x & 31);
  const int row0  = gwave * 16;
  if (row0 >= n) return;
  const int m  = lane & 15;    // A row / B,D column
  const int hh = lane >> 4;    // lane half selects K sub-range
  const float* xrow = x + (size_t)(row0 + m) * DIMF;
  v8f acc = {};
  for (int kb = 0; kb < DIMF; kb += 32) {
    v16bf a, b;
#pragma unroll
    for (int v = 0; v < 8; ++v) {           // A 16x32: lanes0-15 K{0..7,16..23}, lanes16-31 K{8..15,24..31}
      const int k = kb + ((v < 4) ? 0 : 16) + hh * 8 + (v & 3) * 2;
      a[2 * v]     = (__bf16)xrow[k];
      a[2 * v + 1] = (__bf16)xrow[k + 1];
    }
#pragma unroll
    for (int v = 0; v < 8; ++v) {           // B 32x16: lanes0-15 K{0..7,16..23}, lanes16-31 K{8..15,24..31}
      const int k = kb + ((v < 4) ? 0 : 16) + hh * 8 + (v & 3) * 2;
      b[2 * v]     = (__bf16)W1[k * HIDF + m];
      b[2 * v + 1] = (__bf16)W1[(k + 1) * HIDF + m];
    }
    acc = __builtin_amdgcn_wmma_f32_16x16x32_bf16(false, a, false, b, (short)0, acc,
                                                  false, false);
  }
#pragma unroll
  for (int j = 0; j < 8; ++j)               // D: element j -> row j + 8*hh, col m
    h1[(size_t)(row0 + j + 8 * hh) * HIDF + m] = acc[j];
}

// ---------------- GEMM2: lp[N,64] = h2[N,16] @ W2[16,64]  (bf16 WMMA, K padded to 32) --------
__global__ void k_gemm2_wmma(const float* __restrict__ h2, const float* __restrict__ W2,
                             float* __restrict__ lp, int n) {
  const int gwave = (int)((blockIdx.x * blockDim.x + threadIdx.x) >> 5);
  const int lane  = (int)(threadIdx.x & 31);
  const int row0  = gwave * 16;
  if (row0 >= n) return;
  const int m  = lane & 15;
  const int hh = lane >> 4;
  const float* hrow = h2 + (size_t)(row0 + m) * HIDF;
  v16bf a = {};                              // K=16..31 stays zero (padding)
#pragma unroll
  for (int v = 0; v < 4; ++v) {              // real K = hh*8 + 0..7
    const int k = hh * 8 + 2 * v;
    a[2 * v]     = (__bf16)hrow[k];
    a[2 * v + 1] = (__bf16)hrow[k + 1];
  }
#pragma unroll
  for (int ct = 0; ct < 4; ++ct) {           // 4 column tiles of 16 (C=64)
    const int nn = ct * 16 + m;
    v16bf b = {};                            // K=16..31 zero
    {
#pragma unroll
      for (int v = 0; v < 4; ++v) {
        const int k = hh * 8 + 2 * v;        // K = hh*8 + 0..7
        b[2 * v]     = (__bf16)W2[k * CF + nn];
        b[2 * v + 1] = (__bf16)W2[(k + 1) * CF + nn];
      }
    }
    v8f acc = {};
    acc = __builtin_amdgcn_wmma_f32_16x16x32_bf16(false, a, false, b, (short)0, acc,
                                                  false, false);
#pragma unroll
    for (int j = 0; j < 8; ++j)
      lp[(size_t)(row0 + j + 8 * hh) * CF + nn] = acc[j];
  }
}

// ---------------- sparse aggregation ----------------
// Initialize accumulator with the self-loop term: out[i] = h[i] * dinv[i]^2
template <int F>
__global__ void k_selfloop(const float* __restrict__ h, const float* __restrict__ dinv,
                           float* __restrict__ out, int n) {
  long long t = (long long)blockIdx.x * blockDim.x + threadIdx.x;
  long long total = (long long)n * F;
  if (t < total) {
    int node = (int)(t / F);
    float d = dinv[node];
    out[t] = h[t] * d * d;
  }
}

// out[col[e]*F + f] += h[row[e]*F + f] * dinv[row]*dinv[col]  via L2 float atomics.
template <int F>
__global__ void k_agg_edges(const int* __restrict__ row, const int* __restrict__ col,
                            const float* __restrict__ dinv, const float* __restrict__ h,
                            float* out, int E) {
  long long total  = (long long)E * F;
  long long stride = (long long)gridDim.x * blockDim.x;
  for (long long t = (long long)blockIdx.x * blockDim.x + threadIdx.x; t < total; t += stride) {
    int e = (int)(t / F);                    // F is a power of two -> shift
    int f = (int)(t - (long long)e * F);
    int r = row[e], c = col[e];
    float w = dinv[r] * dinv[c];
    atomicAdd(&out[(size_t)c * F + f], h[(size_t)r * F + f] * w);
  }
}

// bias + relu + deterministic dropout(p=0.5, scale 2x), in place
__global__ void k_post1(float* a1, const float* __restrict__ b1, int n) {
  long long t = (long long)blockIdx.x * blockDim.x + threadIdx.x;
  long long total = (long long)n * HIDF;
  if (t < total) {
    int f = (int)(t & (HIDF - 1));
    float v = fmaxf(a1[t] + b1[f], 0.0f);
    unsigned hsh = mix32((unsigned)t ^ 0x9E3779B9u);
    a1[t] = (hsh & 0x80000000u) ? 0.0f : (v * 2.0f);
  }
}

// one wave32 per node: 64 classes = 2 values/lane; shuffle reductions for max & sum
__global__ void k_lsm(float* out, const float* __restrict__ b2, int n) {
  const int gwave = (int)((blockIdx.x * blockDim.x + threadIdx.x) >> 5);
  const int lane  = (int)(threadIdx.x & 31);
  if (gwave >= n) return;
  float* rowp = out + (size_t)gwave * CF;
  float v0 = rowp[lane]      + b2[lane];
  float v1 = rowp[lane + 32] + b2[lane + 32];
  float mx = fmaxf(v0, v1);
#pragma unroll
  for (int off = 16; off; off >>= 1) mx = fmaxf(mx, __shfl_xor(mx, off, 32));
  float s = __expf(v0 - mx) + __expf(v1 - mx);
#pragma unroll
  for (int off = 16; off; off >>= 1) s += __shfl_xor(s, off, 32);
  float lse = mx + __logf(s);
  rowp[lane]      = v0 - lse;
  rowp[lane + 32] = v1 - lse;
}

// ---------------- host orchestration ----------------
extern "C" void kernel_launch(void* const* d_in, const int* in_sizes, int n_in,
                              void* d_out, int out_size, void* d_ws, size_t ws_size,
                              hipStream_t stream) {
  const float* x  = (const float*)d_in[0];
  const int*   ei = (const int*)d_in[1];
  const float* W1 = (const float*)d_in[2];
  const float* b1 = (const float*)d_in[3];
  const float* W2 = (const float*)d_in[4];
  const float* b2 = (const float*)d_in[5];

  const int n = in_sizes[0] / DIMF;          // 100000
  const int E = in_sizes[1] / 2;             // 3.2M
  const int* row = ei;                       // edge_index[0]
  const int* col = ei + E;                   // edge_index[1]

  float* ws   = (float*)d_ws;
  float* dinv = ws;                                // [n]
  float* h1   = dinv + n;                          // [n,16]
  float* a1   = h1 + (size_t)n * HIDF;             // [n,16] -> becomes h2
  float* lp   = a1 + (size_t)n * HIDF;             // [n,64]
  float* outp = (float*)d_out;                     // [n,64]

  const int B = 256;                               // 8 wave32s per block
  // symmetric normalization
  k_deg_init <<<(n + B - 1) / B, B, 0, stream>>>(dinv, n);
  k_deg_accum<<<(E + B - 1) / B, B, 0, stream>>>(col, dinv, E);
  k_rsqrt    <<<(n + B - 1) / B, B, 0, stream>>>(dinv, n);

  // layer 1
  const int waves = (n + 15) / 16;
  k_gemm1_wmma<<<(waves * 32 + B - 1) / B, B, 0, stream>>>(x, W1, h1, n);
  k_selfloop<HIDF><<<(int)(((long long)n * HIDF + B - 1) / B), B, 0, stream>>>(h1, dinv, a1, n);
  k_agg_edges<HIDF><<<8192, B, 0, stream>>>(row, col, dinv, h1, a1, E);
  k_post1<<<(int)(((long long)n * HIDF + B - 1) / B), B, 0, stream>>>(a1, b1, n);

  // layer 2
  k_gemm2_wmma<<<(waves * 32 + B - 1) / B, B, 0, stream>>>(a1, W2, lp, n);
  k_selfloop<CF><<<(int)(((long long)n * CF + B - 1) / B), B, 0, stream>>>(lp, dinv, outp, n);
  k_agg_edges<CF><<<16384, B, 0, stream>>>(row, col, dinv, lp, outp, E);

  // bias + log_softmax (in place on d_out)
  k_lsm<<<(n * 32 + B - 1) / B, B, 0, stream>>>(outp, b2, n);
}
